// GeometricGAT_83184926589450
// MI455X (gfx1250) — matrix operand order repaired and checked by reading, not compiled, not run on hardware
//
#include <hip/hip_runtime.h>
#include <math.h>

typedef __attribute__((ext_vector_type(16))) _Float16 v16h;
typedef __attribute__((ext_vector_type(8)))  _Float16 v8h;
typedef __attribute__((ext_vector_type(8)))  float    v8f;

#define F_IN 128
#define POSD 64
#define TOT  192
#define HC   256   // H*C
#define HEADS 4
#define CH    64
#define APAD 8     // halves of padding per LDS A row (16B) -> conflict-free ds_load_b128

// ---------- helpers ----------
__device__ __forceinline__ unsigned f2ord(float f) {
    unsigned u = __float_as_uint(f);
    return (u & 0x80000000u) ? ~u : (u | 0x80000000u);
}
__device__ __forceinline__ float ord2f(unsigned u) {
    unsigned b = (u & 0x80000000u) ? (u ^ 0x80000000u) : ~u;
    return __uint_as_float(b);
}

// ---------- weight conversion: f32 [K,256] row-major -> packed f16 WMMA B-fragments ----------
// packed[((nt*ksteps + ks)*32 + lane)*16 + j] = W[ks*32 + (lane>>4)*16 + j][nt*16 + (lane&15)]
__global__ void pack_weights_f16(const float* __restrict__ W, _Float16* __restrict__ out, int K) {
    int idx = blockIdx.x * blockDim.x + threadIdx.x;
    int total = K * 256;
    if (idx >= total) return;
    int j    = idx & 15;
    int lane = (idx >> 4) & 31;
    int t    = idx >> 9;
    int ksteps = K >> 5;
    int ks = t % ksteps;
    int nt = t / ksteps;
    int krow = ks * 32 + ((lane >> 4) << 4) + j;
    int col  = nt * 16 + (lane & 15);
    out[idx] = (_Float16)W[(size_t)krow * 256 + col];
}

// ---------- pos encoder + x_comb (f16) ----------
__global__ void node_encode(const float* __restrict__ x, const float* __restrict__ kpts,
                            const float* __restrict__ pts3d,
                            const float* __restrict__ w1, const float* __restrict__ b1,
                            const float* __restrict__ g1, const float* __restrict__ bb1,
                            const float* __restrict__ w2, const float* __restrict__ b2,
                            const float* __restrict__ g2, const float* __restrict__ bb2,
                            _Float16* __restrict__ xcomb, int n) {
    int i = blockIdx.x * blockDim.x + threadIdx.x;
    if (i >= n) return;
    float u = kpts[i * 2 + 0] * (1.0f / 1216.0f);
    float v = kpts[i * 2 + 1] * (1.0f / 352.0f);
    float d = pts3d[i * 3 + 2];
    d = fminf(fmaxf(d, 0.1f), 100.0f);
    float in3[3] = {u, v, d};

    float h1[32];
    float m = 0.f;
    #pragma unroll
    for (int o = 0; o < 32; ++o) {
        float s = b1[o];
        #pragma unroll
        for (int k = 0; k < 3; ++k) s += in3[k] * w1[k * 32 + o];
        h1[o] = s; m += s;
    }
    m *= (1.0f / 32.0f);
    float var = 0.f;
    #pragma unroll
    for (int o = 0; o < 32; ++o) { float t = h1[o] - m; var += t * t; }
    var *= (1.0f / 32.0f);
    float r = rsqrtf(var + 1e-5f);
    #pragma unroll
    for (int o = 0; o < 32; ++o) {
        float t = (h1[o] - m) * r * g1[o] + bb1[o];
        h1[o] = t / (1.0f + expf(-t));           // SiLU
    }

    float h2[64];
    float m2 = 0.f;
    #pragma unroll 4
    for (int o = 0; o < 64; ++o) {
        float s = b2[o];
        #pragma unroll
        for (int k = 0; k < 32; ++k) s += h1[k] * w2[k * 64 + o];
        h2[o] = s; m2 += s;
    }
    m2 *= (1.0f / 64.0f);
    float var2 = 0.f;
    #pragma unroll
    for (int o = 0; o < 64; ++o) { float t = h2[o] - m2; var2 += t * t; }
    var2 *= (1.0f / 64.0f);
    float r2 = rsqrtf(var2 + 1e-5f);

    _Float16* out = xcomb + (size_t)i * TOT;
    #pragma unroll 8
    for (int k = 0; k < F_IN; ++k) {
        float xv = x[(size_t)i * F_IN + k];
        if (xv != xv) xv = 0.f;
        out[k] = (_Float16)xv;
    }
    #pragma unroll
    for (int o = 0; o < 64; ++o) {
        float pv = (h2[o] - m2) * r2 * g2[o] + bb2[o];
        if (pv != pv) pv = 0.f;
        out[F_IN + o] = (_Float16)pv;
    }
}

// ---------- edge attributes -> d_out tail ----------
__global__ void edge_attr_kernel(const int* __restrict__ ei, const float* __restrict__ kpts,
                                 float* __restrict__ ea, int E) {
    int e = blockIdx.x * blockDim.x + threadIdx.x;
    if (e >= E) return;
    int s = ei[e], d = ei[E + e];
    float du = (kpts[d * 2 + 0] - kpts[s * 2 + 0]) * (1.0f / 1216.0f);
    float dv = (kpts[d * 2 + 1] - kpts[s * 2 + 1]) * (1.0f / 352.0f);
    float dist = sqrtf(du * du + dv * dv + 1e-9f);
    ea[(size_t)e * 3 + 0] = du;
    ea[(size_t)e * 3 + 1] = dv;
    ea[(size_t)e * 3 + 2] = dist;
}

// ---------- zero init (accum f32 [N*256], mxkey u32 [N*4], denom f32 [N*4]) ----------
__global__ void zero_init(float* __restrict__ accum, unsigned* __restrict__ mxkey,
                          float* __restrict__ denom, int n) {
    int idx = blockIdx.x * blockDim.x + threadIdx.x;
    int tot = n * HC;
    if (idx < tot) accum[idx] = 0.f;
    if (idx < n * HEADS) { mxkey[idx] = 0u; denom[idx] = 0.f; }
}

// ---------- WMMA GEMM: C[M,256] = A[M,K](f16) * Bpacked(f16) + bias ----------
// A-tile (16 rows x K) staged once per block into LDS (padded rows -> conflict-free
// ds_load_b128 fragment reads); B fragments stream from global/L2 (pre-packed).
__global__ __launch_bounds__(512) void gemm_wmma(const _Float16* __restrict__ A,
                                                 const _Float16* __restrict__ Bp,
                                                 const float* __restrict__ bias,
                                                 float* __restrict__ C, int M, int K) {
    __shared__ _Float16 As[16 * (256 + APAD)];   // max K = 256
    const int tid  = threadIdx.x;
    const int lane = tid & 31;
    const int nt   = tid >> 5;                   // 0..15 : N-tile
    const int row0 = blockIdx.x * 16;
    const int group = lane >> 4;
    const int ksteps = K >> 5;
    const int stride = K + APAD;                 // halves per LDS row

    // cooperative A-tile load: 8 halves (16B) per chunk
    {
        const int chunksPerRow = K >> 3;         // 24 (K=192) or 32 (K=256)
        const int totalChunks  = chunksPerRow << 4;
        if (tid < totalChunks) {
            int row  = tid / chunksPerRow;
            int koff = (tid % chunksPerRow) << 3;
            int grow = row0 + row;
            if (grow >= M) grow = M - 1;         // clamp (stores guarded below)
            v8h val = *(const v8h*)(A + (size_t)grow * K + koff);
            *(v8h*)(&As[row * stride + koff]) = val;
        }
    }
    __syncthreads();

    v8f c = {};
    const _Float16* Arow  = &As[(lane & 15) * stride];
    const _Float16* Bbase = Bp + (size_t)nt * ksteps * 512;
    for (int ks = 0; ks < ksteps; ++ks) {
        v16h a, b;
        const v8h a0 = *(const v8h*)(Arow + ks * 32 + group * 16);
        const v8h a1 = *(const v8h*)(Arow + ks * 32 + group * 16 + 8);
        #pragma unroll
        for (int j = 0; j < 8; ++j) { a[j] = a0[j]; a[8 + j] = a1[j]; }
        const _Float16* bp = Bbase + ((size_t)ks * 32 + lane) * 16;
        #pragma unroll
        for (int j = 0; j < 16; ++j) b[j] = bp[j];
        c = __builtin_amdgcn_wmma_f32_16x16x32_f16(false, a, false, b, (short)0, c, false, false);
    }
    int col = nt * 16 + (lane & 15);
    float bv = bias ? bias[col] : 0.f;
    #pragma unroll
    for (int v = 0; v < 8; ++v) {
        int orow = row0 + v + 8 * group;
        if (orow < M) C[(size_t)orow * HC + col] = c[v] + bv;
    }
}

// ---------- edge pass 1: logits + segment max (wave per edge, 8 ch/lane) ----------
__global__ void edge_logits_kernel(const int* __restrict__ ei, const float* __restrict__ ea,
                                   const float* __restrict__ xl, const float* __restrict__ xr,
                                   const float* __restrict__ We, const float* __restrict__ att,
                                   float* __restrict__ logits, unsigned* __restrict__ mxkey, int E) {
    int e = (int)((blockIdx.x * (size_t)blockDim.x + threadIdx.x) >> 5);
    int lane = threadIdx.x & 31;
    if (e >= E) return;
    int src = ei[e], dst = ei[E + e];
    float e0 = ea[(size_t)e * 3 + 0], e1 = ea[(size_t)e * 3 + 1], e2 = ea[(size_t)e * 3 + 2];
    int c0 = lane * 8;
    const float* xls = xl + (size_t)src * HC + c0;
    const float* xrd = xr + (size_t)dst * HC + c0;
    float p = 0.f;
    #pragma unroll
    for (int t = 0; t < 8; ++t) {
        int c = c0 + t;
        float ee = e0 * We[c] + e1 * We[256 + c] + e2 * We[512 + c];
        float z = xls[t] + xrd[t] + ee;
        z = (z > 0.f) ? z : 0.2f * z;            // leaky_relu(0.2)
        p += att[c] * z;
    }
    p += __shfl_xor(p, 1);
    p += __shfl_xor(p, 2);
    p += __shfl_xor(p, 4);
    if ((lane & 7) == 0) {
        int h = lane >> 3;
        logits[(size_t)e * HEADS + h] = p;
        atomicMax(mxkey + (size_t)dst * HEADS + h, f2ord(p));
    }
}

// ---------- edge pass 2: exp + denom ----------
__global__ void edge_exp_kernel(const int* __restrict__ ei, float* __restrict__ logits,
                                const unsigned* __restrict__ mxkey, float* __restrict__ denom, int E) {
    int idx = blockIdx.x * blockDim.x + threadIdx.x;
    if (idx >= E * HEADS) return;
    int e = idx >> 2, h = idx & 3;
    int dst = ei[E + e];
    float mx = ord2f(mxkey[(size_t)dst * HEADS + h]);
    float a = expf(logits[idx] - mx);
    logits[idx] = a;
    atomicAdd(denom + (size_t)dst * HEADS + h, a);
}

// ---------- edge pass 3: scatter messages ----------
__global__ void edge_scatter_kernel(const int* __restrict__ ei, const float* __restrict__ logits,
                                    const float* __restrict__ denom, const float* __restrict__ xl,
                                    float* __restrict__ accum, int E) {
    int e = (int)((blockIdx.x * (size_t)blockDim.x + threadIdx.x) >> 5);
    int lane = threadIdx.x & 31;
    if (e >= E) return;
    int src = ei[e], dst = ei[E + e];
    int h = lane >> 3;
    float alpha = logits[(size_t)e * HEADS + h] / (denom[(size_t)dst * HEADS + h] + 1e-16f);
    int c0 = lane * 8;
    const float* xls = xl + (size_t)src * HC + c0;
    float* acc = accum + (size_t)dst * HC + c0;
    #pragma unroll
    for (int t = 0; t < 8; ++t) atomicAdd(acc + t, xls[t] * alpha);
}

// ---------- post: +bias -> LN -> +res -> SiLU -> LN -> f16 (wave per node) ----------
__global__ void node_post_kernel(const float* __restrict__ accum, const float* __restrict__ conv_b,
                                 const float* __restrict__ ng, const float* __restrict__ nb,
                                 const float* __restrict__ res,
                                 const float* __restrict__ pg, const float* __restrict__ pb,
                                 _Float16* __restrict__ outh, int n) {
    int node = (int)((blockIdx.x * (size_t)blockDim.x + threadIdx.x) >> 5);
    int lane = threadIdx.x & 31;
    if (node >= n) return;
    int c0 = lane * 8;
    float v[8];
    float s = 0.f, s2 = 0.f;
    #pragma unroll
    for (int t = 0; t < 8; ++t) {
        float x = accum[(size_t)node * HC + c0 + t] + conv_b[c0 + t];
        v[t] = x; s += x; s2 += x * x;
    }
    #pragma unroll
    for (int m = 1; m < 32; m <<= 1) { s += __shfl_xor(s, m); s2 += __shfl_xor(s2, m); }
    float mean = s * (1.0f / 256.0f);
    float var = s2 * (1.0f / 256.0f) - mean * mean;
    float r = rsqrtf(var + 1e-5f);
    float s_ = 0.f, s2_ = 0.f;
    #pragma unroll
    for (int t = 0; t < 8; ++t) {
        int c = c0 + t;
        float y = (v[t] - mean) * r * ng[c] + nb[c] + res[(size_t)node * HC + c];
        y = y / (1.0f + expf(-y));               // SiLU
        v[t] = y; s_ += y; s2_ += y * y;
    }
    #pragma unroll
    for (int m = 1; m < 32; m <<= 1) { s_ += __shfl_xor(s_, m); s2_ += __shfl_xor(s2_, m); }
    float mean2 = s_ * (1.0f / 256.0f);
    float var2 = s2_ * (1.0f / 256.0f) - mean2 * mean2;
    float r2 = rsqrtf(var2 + 1e-5f);
    #pragma unroll
    for (int t = 0; t < 8; ++t) {
        int c = c0 + t;
        outh[(size_t)node * HC + c] = (_Float16)((v[t] - mean2) * r2 * pg[c] + pb[c]);
    }
}

// ---------- host ----------
extern "C" void kernel_launch(void* const* d_in, const int* in_sizes, int n_in,
                              void* d_out, int out_size, void* d_ws, size_t ws_size,
                              hipStream_t stream) {
    const float* x      = (const float*)d_in[0];
    const int*   ei     = (const int*)d_in[1];
    const float* kpts   = (const float*)d_in[2];
    const float* pts3d  = (const float*)d_in[3];
    const float* pe_w1  = (const float*)d_in[4];
    const float* pe_b1  = (const float*)d_in[5];
    const float* pe_g1  = (const float*)d_in[6];
    const float* pe_bb1 = (const float*)d_in[7];
    const float* pe_w2  = (const float*)d_in[8];
    const float* pe_b2  = (const float*)d_in[9];
    const float* pe_g2  = (const float*)d_in[10];
    const float* pe_bb2 = (const float*)d_in[11];
    const float* lin_l_w = (const float*)d_in[12];
    const float* lin_l_b = (const float*)d_in[13];
    const float* lin_r_w = (const float*)d_in[14];
    const float* lin_r_b = (const float*)d_in[15];
    const float* lin_e_w = (const float*)d_in[16];
    const float* att     = (const float*)d_in[17];
    const float* conv_b  = (const float*)d_in[18];
    const float* res_w   = (const float*)d_in[19];
    const float* res_b   = (const float*)d_in[20];
    const float* norm_g  = (const float*)d_in[21];
    const float* norm_b  = (const float*)d_in[22];
    const float* post_g  = (const float*)d_in[23];
    const float* post_b  = (const float*)d_in[24];
    const float* proj_w  = (const float*)d_in[25];
    const float* proj_b  = (const float*)d_in[26];

    const int N = in_sizes[2] / 2;          // kpts [N,2]
    const int E = in_sizes[1] / 2;          // edge_index [2,E]

    float* out_nodes = (float*)d_out;                       // [N,256]
    float* out_edges = (float*)d_out + (size_t)N * HC;      // [E,3]

    // workspace carving
    size_t off = 0;
    auto carve = [&](size_t bytes) -> char* {
        char* p = (char*)d_ws + off;
        off += (bytes + 255) & ~size_t(255);
        return p;
    };
    _Float16* xcomb_h = (_Float16*)carve((size_t)N * TOT * 2);
    _Float16* Wl_p    = (_Float16*)carve((size_t)TOT * 256 * 2);
    _Float16* Wr_p    = (_Float16*)carve((size_t)TOT * 256 * 2);
    _Float16* Wres_p  = (_Float16*)carve((size_t)TOT * 256 * 2);
    _Float16* Wproj_p = (_Float16*)carve((size_t)256 * 256 * 2);
    float*    xl      = (float*)carve((size_t)N * HC * 4);
    float*    xr      = (float*)carve((size_t)N * HC * 4);
    float*    resbuf  = (float*)carve((size_t)N * HC * 4);
    float*    logits  = (float*)carve((size_t)E * HEADS * 4);
    unsigned* mxkey   = (unsigned*)carve((size_t)N * HEADS * 4);
    float*    denom   = (float*)carve((size_t)N * HEADS * 4);
    float*    accum   = (float*)carve((size_t)N * HC * 4);
    _Float16* post_h  = (_Float16*)carve((size_t)N * HC * 2);
    (void)ws_size; (void)n_in; (void)out_size;

    // 1) pack weights -> f16 WMMA fragments
    {
        int tot = TOT * 256;
        pack_weights_f16<<<(tot + 255) / 256, 256, 0, stream>>>(lin_l_w, Wl_p, TOT);
        pack_weights_f16<<<(tot + 255) / 256, 256, 0, stream>>>(lin_r_w, Wr_p, TOT);
        pack_weights_f16<<<(tot + 255) / 256, 256, 0, stream>>>(res_w,  Wres_p, TOT);
        int tot2 = 256 * 256;
        pack_weights_f16<<<(tot2 + 255) / 256, 256, 0, stream>>>(proj_w, Wproj_p, 256);
    }
    // 2) pos encoder + x_comb
    node_encode<<<(N + 255) / 256, 256, 0, stream>>>(x, kpts, pts3d,
        pe_w1, pe_b1, pe_g1, pe_bb1, pe_w2, pe_b2, pe_g2, pe_bb2, xcomb_h, N);
    // 3) edge attrs (straight into output tail)
    edge_attr_kernel<<<(E + 255) / 256, 256, 0, stream>>>(ei, kpts, out_edges, E);
    // 4) zero accumulators
    zero_init<<<((size_t)N * HC + 255) / 256, 256, 0, stream>>>(accum, mxkey, denom, N);
    // 5) node GEMMs via WMMA (A staged through LDS)
    int mblocks = (N + 15) / 16;
    gemm_wmma<<<mblocks, 512, 0, stream>>>(xcomb_h, Wl_p,   lin_l_b, xl,     N, TOT);
    gemm_wmma<<<mblocks, 512, 0, stream>>>(xcomb_h, Wr_p,   lin_r_b, xr,     N, TOT);
    gemm_wmma<<<mblocks, 512, 0, stream>>>(xcomb_h, Wres_p, res_b,   resbuf, N, TOT);
    // 6-8) attention (wave per edge)
    {
        size_t ethreads = (size_t)E * 32;
        int eblocks = (int)((ethreads + 255) / 256);
        edge_logits_kernel<<<eblocks, 256, 0, stream>>>(ei, out_edges, xl, xr, lin_e_w, att,
                                                        logits, mxkey, E);
        edge_exp_kernel<<<((size_t)E * HEADS + 255) / 256, 256, 0, stream>>>(ei, logits, mxkey, denom, E);
        edge_scatter_kernel<<<eblocks, 256, 0, stream>>>(ei, logits, denom, xl, accum, E);
    }
    // 9) LN -> +res -> SiLU -> LN -> f16
    {
        size_t nthreads = (size_t)N * 32;
        node_post_kernel<<<(nthreads + 255) / 256, 256, 0, stream>>>(accum, conv_b, norm_g, norm_b,
                                                                     resbuf, post_g, post_b, post_h, N);
    }
    // 10) proj GEMM -> d_out
    gemm_wmma<<<mblocks, 512, 0, stream>>>(post_h, Wproj_p, proj_b, out_nodes, N, 256);
}
